// CNNROIMapEncoder_40269613367566
// MI455X (gfx1250) — matrix operand order repaired and checked by reading, not compiled, not run on hardware
//
#include <hip/hip_runtime.h>
#include <hip/hip_bf16.h>
#include <math.h>

// ---------------------------------------------------------------------------
// CDNA5 (gfx1250) types
// ---------------------------------------------------------------------------
typedef __attribute__((ext_vector_type(16))) __bf16 v16bf;
typedef __attribute__((ext_vector_type(8)))  __bf16 v8bf;
typedef __attribute__((ext_vector_type(8)))  float  v8f;
typedef unsigned int u32x4 __attribute__((ext_vector_type(4)));
typedef int          i32x4 __attribute__((ext_vector_type(4)));
typedef int          i32x8 __attribute__((ext_vector_type(8)));

#define LEAK 0.2f

// Tensor Data Mover availability (device pass only; host pass takes fallback)
#if defined(__HIP_DEVICE_COMPILE__) && __has_builtin(__builtin_amdgcn_tensor_load_to_lds)
#define USE_TDM 1
#else
#define USE_TDM 0
#endif
#if __has_include(<hip/amd_detail/amd_gfx1250_TDM.h>)
#define TDM_ARITY6 1
#else
#define TDM_ARITY6 0
#endif

#if USE_TDM
// 2D bf16 tile load Global->LDS via TDM.
//   tile = tile_d1 rows x 32 bf16 (64B) per row, row stride = stride_el elems.
//   LDS rows padded 64B+16B (pad_interval code 3 = 16 DW, pad_amount code 3 =
//   4 DW) -> 40 bf16/row. tensor_d0/d1 = extents REMAINING from the tile
//   origin so TDM OOB reads return zero (handles the conv1 K=75 tail).
__device__ __forceinline__ void tdm_load_tile_bf16(
    unsigned lds_byte_addr, const void* gptr,
    unsigned tensor_d0, unsigned tensor_d1,
    unsigned long long stride_el, unsigned tile_d1) {
  size_t ga = (size_t)gptr;
  u32x4 g0;
  g0[0] = 1u;                                     // count=1 valid user D#
  g0[1] = lds_byte_addr;                          // lds_addr
  g0[2] = (unsigned)(ga & 0xffffffffu);           // global_addr[31:0]
  g0[3] = (unsigned)((ga >> 32) & 0x01ffffffu) | (2u << 30);  // addr[56:32], type=2
  i32x8 g1;
  unsigned w0 = (1u << 16)                        // data_size = 1 -> 2 bytes
              | (1u << 20)                        // pad_enable
              | (3u << 22)                        // pad_interval: 16 DWORDs
              | (3u << 25);                       // pad_amount: 4 DWORDs
  g1[0] = (int)w0;
  g1[1] = (int)((tensor_d0 & 0xffffu) << 16);                 // dim0[15:0]
  g1[2] = (int)(((tensor_d0 >> 16) & 0xffffu) | ((tensor_d1 & 0xffffu) << 16));
  g1[3] = (int)(((tensor_d1 >> 16) & 0xffffu) | (32u << 16)); // tile_dim0 = 32
  g1[4] = (int)(tile_d1 & 0xffffu);                           // tile_dim1, tile_dim2=0
  g1[5] = (int)(unsigned)(stride_el & 0xffffffffu);           // dim0_stride[31:0]
  g1[6] = (int)(unsigned)((stride_el >> 32) & 0xffffu);       // dim0_stride[47:32]
  g1[7] = 0;                                                  // dim1_stride unused (2D)
  i32x4 z4 = {0, 0, 0, 0};
#if TDM_ARITY6
  i32x8 z8 = {0, 0, 0, 0, 0, 0, 0, 0};
  __builtin_amdgcn_tensor_load_to_lds(g0, g1, z4, z4, z8, 0);
#else
  __builtin_amdgcn_tensor_load_to_lds(g0, g1, z4, z4, 0);
#endif
}
#endif  // USE_TDM

__device__ __forceinline__ unsigned lds_addr_of(const void* p) {
  // LDS aperture: flat addr[31:0] == LDS byte offset (ISA 10.2)
  return (unsigned)(size_t)p;
}

// Fragment loads from natural-K-order LDS rows (stride 40 bf16 = 80 B, 16B aligned)
// A 16x32 bf16 frag, lane(l16=m, half h): K = {8h..8h+7} ++ {16+8h..23+8h}
__device__ __forceinline__ v16bf load_a_frag(const __bf16* row, int half) {
  v8bf lo = *(const v8bf*)(row + half * 8);
  v8bf hi = *(const v8bf*)(row + 16 + half * 8);
  v16bf f;
#pragma unroll
  for (int e = 0; e < 8; ++e) { f[e] = lo[e]; f[e + 8] = hi[e]; }
  return f;
}
// B 32x16 bf16 frag, lane(l16=n, half h): K = {8h..8h+7} ++ {16+8h..23+8h} (rows stored [n][k])
__device__ __forceinline__ v16bf load_b_frag(const __bf16* row, int half) {
  v8bf lo = *(const v8bf*)(row + half * 8);
  v8bf hi = *(const v8bf*)(row + 16 + half * 8);
  v16bf f;
#pragma unroll
  for (int e = 0; e < 8; ++e) { f[e] = lo[e]; f[e + 8] = hi[e]; }
  return f;
}

#define WMMA_BF16(A, Bf, C) \
  __builtin_amdgcn_wmma_f32_16x16x32_bf16(false, (A), false, (Bf), (short)0, (C), false, false)

// ---------------------------------------------------------------------------
// fp32 -> bf16 cast
// ---------------------------------------------------------------------------
__global__ void cast_f32_bf16_kernel(const float* __restrict__ src,
                                     __bf16* __restrict__ dst, int n) {
  int i = blockIdx.x * blockDim.x + threadIdx.x;
  int stride = gridDim.x * blockDim.x;
  for (; i < n; i += stride) dst[i] = (__bf16)src[i];
}

// ---------------------------------------------------------------------------
// Implicit-GEMM conv + bias + leaky-relu. M=Cout, N=B*OH*OW, K=Cin*KH*KW.
// 8 waves (WM x WN), each wave owns a 32x32 C tile (2x2 WMMA subtiles).
// Weight (A) tile: TDM, double-buffered, pipelined one K-step ahead
// (s_wait_tensorcnt <= in-flight). im2col (B) tile: manual gather with the
// pixel decode hoisted out of the K loop (single buffer, inside barrier pair).
// ---------------------------------------------------------------------------
template <typename TIN, int CIN, int COUT, int KH, int KW, int STRIDE, int PAD,
          int HIN, int WIN, int HOUT, int WOUT, int WM, int WN>
__global__ __launch_bounds__(256) void conv_wmma_kernel(
    const TIN* __restrict__ in,        // [B][CIN][HIN][WIN]
    const __bf16* __restrict__ wt,     // [COUT][K_TOTAL] bf16
    const float* __restrict__ bias,    // [COUT]
    __bf16* __restrict__ out,          // [B][COUT][HOUT][WOUT]
    int B) {
  constexpr int M_TILE = WM * 32;
  constexpr int N_TILE = WN * 32;
  constexpr int K_TOTAL = CIN * KH * KW;
  constexpr int KSTEPS = (K_TOTAL + 31) / 32;
  constexpr int KKSTEP = 256 / N_TILE;   // N_TILE is 128 or 256

  __shared__ alignas(16) __bf16 As[2][M_TILE][40];   // [buf][m][k]
  __shared__ alignas(16) __bf16 Bs[N_TILE][40];      // [n][k] transposed im2col

  const int Ntot  = B * HOUT * WOUT;
  const int Nbase = blockIdx.x * N_TILE;
  const int Mbase = blockIdx.y * M_TILE;
  const int tid   = threadIdx.x;
  const int lane  = tid & 31;
  const int wave  = tid >> 5;
  const int wm    = wave % WM;
  const int wn    = wave / WM;
  const int half  = lane >> 4;
  const int l16   = lane & 15;

  // ---- loop-invariant im2col pixel decode (n fixed per thread) ----
  const int n_fixed = tid & (N_TILE - 1);
  const int kk_base = tid / N_TILE;
  const int p_fix   = Nbase + n_fixed;
  const bool pvalid = p_fix < Ntot;
  int ohs = 0, ows = 0;
  const TIN* inb = in;
  if (pvalid) {
    int bb = p_fix / (HOUT * WOUT), pr = p_fix % (HOUT * WOUT);
    int oh = pr / WOUT, ow = pr % WOUT;
    ohs = oh * STRIDE - PAD;
    ows = ow * STRIDE - PAD;
    inb = in + (size_t)bb * CIN * HIN * WIN;
  }

  v8f acc[2][2] = {};

#if USE_TDM
  if (tid == 0) {  // prologue: stream first weight tile
    tdm_load_tile_bf16(lds_addr_of(&As[0][0][0]),
                       wt + (size_t)Mbase * K_TOTAL,
                       (unsigned)K_TOTAL, (unsigned)(COUT - Mbase),
                       (unsigned long long)K_TOTAL, M_TILE);
  }
#endif

  for (int ks = 0; ks < KSTEPS; ++ks) {
    const int cur = ks & 1;
    const int k0 = ks * 32;
#if USE_TDM
    if (tid == 0 && ks + 1 < KSTEPS) {  // stream NEXT weight tile into alt buffer
      tdm_load_tile_bf16(lds_addr_of(&As[cur ^ 1][0][0]),
                         wt + (size_t)Mbase * K_TOTAL + (k0 + 32),
                         (unsigned)(K_TOTAL - (k0 + 32)), (unsigned)(COUT - Mbase),
                         (unsigned long long)K_TOTAL, M_TILE);
    }
#else
    for (int idx = tid; idx < M_TILE * 32; idx += 256) {
      int m = idx >> 5, kk = idx & 31, k = k0 + kk;
      As[cur][m][kk] = (k < K_TOTAL) ? wt[(Mbase + m) * K_TOTAL + k] : (__bf16)0.f;
    }
#endif
    // im2col gather for this K-slice (only k -> (ci,kh,kw) decoded per element)
#pragma unroll
    for (int kk = kk_base; kk < 32; kk += KKSTEP) {
      int k = k0 + kk;
      float v = 0.f;
      if (pvalid && k < K_TOTAL) {
        int ci = k / (KH * KW), rem = k % (KH * KW);
        int kh = rem / KW, kw = rem % KW;
        int ih = ohs + kh, iw = ows + kw;
        if (ih >= 0 && ih < HIN && iw >= 0 && iw < WIN)
          v = (float)inb[(ci * HIN + ih) * WIN + iw];
      }
      Bs[n_fixed][kk] = (__bf16)v;
    }
#if USE_TDM
    // current buffer resident when <= in-flight(next) remain outstanding
    if (tid == 0) {
      if (ks + 1 < KSTEPS) __builtin_amdgcn_s_wait_tensorcnt(1);
      else                 __builtin_amdgcn_s_wait_tensorcnt(0);
    }
#endif
    __syncthreads();

    const __bf16* arow0 = &As[cur][wm * 32 + l16][0];
    const __bf16* brow0 = &Bs[wn * 32 + l16][0];
    v16bf a0 = load_a_frag(arow0, half);
    v16bf a1 = load_a_frag(arow0 + 16 * 40, half);
    v16bf b0 = load_b_frag(brow0, half);
    v16bf b1 = load_b_frag(brow0 + 16 * 40, half);
    acc[0][0] = WMMA_BF16(a0, b0, acc[0][0]);
    acc[0][1] = WMMA_BF16(a0, b1, acc[0][1]);
    acc[1][0] = WMMA_BF16(a1, b0, acc[1][0]);
    acc[1][1] = WMMA_BF16(a1, b1, acc[1][1]);
    __syncthreads();  // protects As[cur^1]/Bs reuse next iteration
  }

  // epilogue: C 16x16 f32 layout (lane l16 = N, vgpr vv -> M = vv + 8*half)
#pragma unroll
  for (int sj = 0; sj < 2; ++sj) {
    const int p = Nbase + wn * 32 + sj * 16 + l16;
    if (p < Ntot) {
      int b = p / (HOUT * WOUT), pr = p % (HOUT * WOUT);
      int oh = pr / WOUT, ow = pr % WOUT;
#pragma unroll
      for (int si = 0; si < 2; ++si) {
#pragma unroll
        for (int vv = 0; vv < 8; ++vv) {
          int c = Mbase + wm * 32 + si * 16 + half * 8 + vv;
          float val = acc[si][sj][vv] + bias[c];
          val = val > 0.f ? val : LEAK * val;
          out[((b * COUT + c) * HOUT + oh) * WOUT + ow] = (__bf16)val;
        }
      }
    }
  }
}

// ---------------------------------------------------------------------------
// ROI rotated-grid bilinear sampling (reference clip/step semantics exactly).
// feat [B][128][56][56] bf16 -> outflat [B*R][128*100] bf16
// ---------------------------------------------------------------------------
__global__ void roi_kernel(const __bf16* __restrict__ feat,
                           const float* __restrict__ ROI,  // [B][R][7]
                           __bf16* __restrict__ outflat, int B, int R) {
  int idx = blockIdx.x * blockDim.x + threadIdx.x;
  int total = B * R * 100;
  if (idx >= total) return;
  int j = idx % 10, t = idx / 10;
  int i = t % 10; t /= 10;
  int r = t % R;  int b = t / R;

  const float* roi = ROI + (size_t)(b * R + r) * 7;
  float cx = roi[0], cy = roi[1];
  float W1 = roi[2], W2 = roi[3], H1 = roi[4], H2 = roi[5], psi = roi[6];
  float WHx = W1 + W2, WHy = H1 + H2;
  float ofx = (W1 - W2) * 0.5f, ofy = (H1 - H2) * 0.5f;
  float s = sinf(psi), c = cosf(psi);
  float gx = ((float)j - 4.5f) * 0.1f;
  float gy = ((float)i - 4.5f) * 0.1f;
  float gix = gx * WHx - ofx, giy = gy * WHy - ofy;
  float xs = gix * c + giy * s + cx;
  float ys = -gix * s + giy * c + cy;

  int x0 = (int)floorf(xs), x1 = x0 + 1;
  int y0 = (int)floorf(ys), y1 = y0 + 1;
  x0 = min(max(x0, 0), 55); x1 = min(max(x1, 0), 55);
  y0 = min(max(y0, 0), 55); y1 = min(max(y1, 0), 55);
  float x0f = (float)x0, x1f = (float)x1, y0f = (float)y0, y1f = (float)y1;
  float step = fminf(fmaxf((x1f - x0f) * (y1f - y0f), 0.001f), 2.0f);
  float inv = 1.0f / step;
  float wa = (x1f - xs) * (y1f - ys) * inv;
  float wb = (x1f - xs) * (ys - y0f) * inv;
  float wc = (xs - x0f) * (y1f - ys) * inv;
  float wd = (xs - x0f) * (ys - y0f) * inv;

  const __bf16* fb = feat + (size_t)b * 128 * 3136;
  int oa = y0 * 56 + x0, ob = y1 * 56 + x0, oc = y0 * 56 + x1, od = y1 * 56 + x1;
  __bf16* outp = outflat + (size_t)(b * R + r) * 12800 + i * 10 + j;
#pragma unroll 4
  for (int ch = 0; ch < 128; ++ch) {
    const __bf16* f = fb + ch * 3136;
    float val = (float)f[oa] * wa + (float)f[ob] * wb +
                (float)f[oc] * wc + (float)f[od] * wd;
    outp[ch * 100] = (__bf16)val;
  }
}

// ---------------------------------------------------------------------------
// FC: out[M][N] = A[M][K] * W[N][K]^T + bias.
// Both tiles TDM-streamed, double-buffered, pipelined one K-step ahead.
// ---------------------------------------------------------------------------
template <int K_TOTAL, int NCOL, int WM, int WN>
__global__ __launch_bounds__(256) void fc_wmma_kernel(
    const __bf16* __restrict__ A,     // [Mtot][K_TOTAL]
    const __bf16* __restrict__ Wt,    // [NCOL][K_TOTAL]
    const float* __restrict__ bias,   // [NCOL]
    float* __restrict__ out,          // [Mtot][NCOL]
    int Mtot) {
  constexpr int M_TILE = WM * 32;
  constexpr int N_TILE = WN * 32;
  constexpr int KSTEPS = K_TOTAL / 32;
  __shared__ alignas(16) __bf16 As[2][M_TILE][40];
  __shared__ alignas(16) __bf16 Bs[2][N_TILE][40];

  const int Mbase = blockIdx.x * M_TILE;
  const int Nbase = blockIdx.y * N_TILE;
  const int tid = threadIdx.x;
  const int lane = tid & 31;
  const int wave = tid >> 5;
  const int wm = wave % WM;
  const int wn = wave / WM;
  const int half = lane >> 4;
  const int l16 = lane & 15;

  v8f acc[2][2] = {};

#if USE_TDM
  if (tid == 0) {  // prologue
    tdm_load_tile_bf16(lds_addr_of(&As[0][0][0]), A + (size_t)Mbase * K_TOTAL,
                       (unsigned)K_TOTAL, (unsigned)(Mtot - Mbase),
                       (unsigned long long)K_TOTAL, M_TILE);
    tdm_load_tile_bf16(lds_addr_of(&Bs[0][0][0]), Wt + (size_t)Nbase * K_TOTAL,
                       (unsigned)K_TOTAL, (unsigned)(NCOL - Nbase),
                       (unsigned long long)K_TOTAL, N_TILE);
  }
#endif

  for (int ks = 0; ks < KSTEPS; ++ks) {
    const int cur = ks & 1;
    const int k0 = ks * 32;
#if USE_TDM
    if (tid == 0 && ks + 1 < KSTEPS) {
      tdm_load_tile_bf16(lds_addr_of(&As[cur ^ 1][0][0]),
                         A + (size_t)Mbase * K_TOTAL + (k0 + 32),
                         (unsigned)(K_TOTAL - (k0 + 32)), (unsigned)(Mtot - Mbase),
                         (unsigned long long)K_TOTAL, M_TILE);
      tdm_load_tile_bf16(lds_addr_of(&Bs[cur ^ 1][0][0]),
                         Wt + (size_t)Nbase * K_TOTAL + (k0 + 32),
                         (unsigned)(K_TOTAL - (k0 + 32)), (unsigned)(NCOL - Nbase),
                         (unsigned long long)K_TOTAL, N_TILE);
    }
    if (tid == 0) {
      if (ks + 1 < KSTEPS) __builtin_amdgcn_s_wait_tensorcnt(2);
      else                 __builtin_amdgcn_s_wait_tensorcnt(0);
    }
#else
    for (int idx = tid; idx < M_TILE * 32; idx += 256) {
      int m = idx >> 5, kk = idx & 31;
      int row = Mbase + m;
      As[cur][m][kk] = (row < Mtot) ? A[(size_t)row * K_TOTAL + k0 + kk] : (__bf16)0.f;
    }
    for (int idx = tid; idx < 32 * N_TILE; idx += 256) {
      int kk = idx / N_TILE, n = idx % N_TILE;
      Bs[cur][n][kk] = Wt[(size_t)(Nbase + n) * K_TOTAL + k0 + kk];
    }
#endif
    __syncthreads();

    const __bf16* arow0 = &As[cur][wm * 32 + l16][0];
    const __bf16* brow0 = &Bs[cur][wn * 32 + l16][0];
    v16bf a0 = load_a_frag(arow0, half);
    v16bf a1 = load_a_frag(arow0 + 16 * 40, half);
    v16bf b0 = load_b_frag(brow0, half);
    v16bf b1 = load_b_frag(brow0 + 16 * 40, half);
    acc[0][0] = WMMA_BF16(a0, b0, acc[0][0]);
    acc[0][1] = WMMA_BF16(a0, b1, acc[0][1]);
    acc[1][0] = WMMA_BF16(a1, b0, acc[1][0]);
    acc[1][1] = WMMA_BF16(a1, b1, acc[1][1]);
    __syncthreads();
  }

#pragma unroll
  for (int sj = 0; sj < 2; ++sj) {
    const int col = Nbase + wn * 32 + sj * 16 + l16;
#pragma unroll
    for (int si = 0; si < 2; ++si) {
#pragma unroll
      for (int vv = 0; vv < 8; ++vv) {
        int row = Mbase + wm * 32 + si * 16 + half * 8 + vv;
        if (row < Mtot) out[(size_t)row * NCOL + col] = acc[si][sj][vv] + bias[col];
      }
    }
  }
}

// ---------------------------------------------------------------------------
// Host launcher
// ---------------------------------------------------------------------------
extern "C" void kernel_launch(void* const* d_in, const int* in_sizes, int n_in,
                              void* d_out, int out_size, void* d_ws, size_t ws_size,
                              hipStream_t stream) {
  const float* x    = (const float*)d_in[0];   // [B][3][224][224]
  const float* ROI  = (const float*)d_in[1];   // [B][R][7]
  const float* w1   = (const float*)d_in[2];
  const float* b1   = (const float*)d_in[3];
  const float* w2   = (const float*)d_in[4];
  const float* b2   = (const float*)d_in[5];
  const float* w3   = (const float*)d_in[6];
  const float* b3   = (const float*)d_in[7];
  const float* fcw  = (const float*)d_in[8];   // [256][12800]
  const float* fcb  = (const float*)d_in[9];

  const int B = in_sizes[0] / (3 * 224 * 224);
  const int R = in_sizes[1] / (B * 7);

  char* ws = (char*)d_ws;
  size_t off = 0;
  auto alloc_bf16 = [&](size_t elems) {
    __bf16* p = (__bf16*)(ws + off);
    off += ((elems * sizeof(__bf16) + 255) / 256) * 256;
    return p;
  };
  __bf16* w1b     = alloc_bf16(32 * 75);
  __bf16* w2b     = alloc_bf16(64 * 800);
  __bf16* w3b     = alloc_bf16(128 * 576);
  __bf16* fcwb    = alloc_bf16((size_t)256 * 12800);
  __bf16* act1    = alloc_bf16((size_t)B * 32 * 112 * 112);
  __bf16* act2    = alloc_bf16((size_t)B * 64 * 56 * 56);
  __bf16* act3    = alloc_bf16((size_t)B * 128 * 56 * 56);
  __bf16* roiflat = alloc_bf16((size_t)B * R * 12800);

  cast_f32_bf16_kernel<<<16, 256, 0, stream>>>(w1, w1b, 32 * 75);
  cast_f32_bf16_kernel<<<64, 256, 0, stream>>>(w2, w2b, 64 * 800);
  cast_f32_bf16_kernel<<<96, 256, 0, stream>>>(w3, w3b, 128 * 576);
  cast_f32_bf16_kernel<<<2048, 256, 0, stream>>>(fcw, fcwb, 256 * 12800);

  // conv1: 3->32, 5x5, s2, p2, 224->112   (M_TILE=32, N_TILE=256)
  {
    int Ntot = B * 112 * 112;
    dim3 grid((Ntot + 255) / 256, 1);
    conv_wmma_kernel<float, 3, 32, 5, 5, 2, 2, 224, 224, 112, 112, 1, 8>
        <<<grid, 256, 0, stream>>>(x, w1b, b1, act1, B);
  }
  // conv2: 32->64, 5x5, s2, p2, 112->56   (M_TILE=64, N_TILE=128)
  {
    int Ntot = B * 56 * 56;
    dim3 grid((Ntot + 127) / 128, 1);
    conv_wmma_kernel<__bf16, 32, 64, 5, 5, 2, 2, 112, 112, 56, 56, 2, 4>
        <<<grid, 256, 0, stream>>>(act1, w2b, b2, act2, B);
  }
  // conv3: 64->128, 3x3, s1, p1, 56->56   (M_TILE=64 x2 blocks, N_TILE=128)
  {
    int Ntot = B * 56 * 56;
    dim3 grid((Ntot + 127) / 128, 2);
    conv_wmma_kernel<__bf16, 64, 128, 3, 3, 1, 1, 56, 56, 56, 56, 2, 4>
        <<<grid, 256, 0, stream>>>(act2, w3b, b3, act3, B);
  }
  // ROI bilinear sampling
  roi_kernel<<<(B * R * 100 + 255) / 256, 256, 0, stream>>>(act3, ROI, roiflat, B, R);

  // FC: [B*R, 12800] @ [256, 12800]^T + bias -> d_out (fp32)
  {
    int Mtot = B * R;
    dim3 grid((Mtot + 63) / 64, 256 / 128);
    fc_wmma_kernel<12800, 256, 2, 4>
        <<<grid, 256, 0, stream>>>(roiflat, fcwb, fcb, (float*)d_out, Mtot);
  }
}
